// Transformer_noGLULayer_50783693307976
// MI455X (gfx1250) — matrix operand order repaired and checked by reading, not compiled, not run on hardware
//
#include <hip/hip_runtime.h>
#include <stdint.h>

// ---------------------------------------------------------------------------
// CDNA5 (gfx1250) transformer attention layer.
// bf16 WMMA (v_wmma_f32_16x16x32_bf16) for all four GEMM stages +
// async global->LDS double-buffered tile staging (ASYNCcnt pipeline).
// wave32 throughout.
// ---------------------------------------------------------------------------

typedef __attribute__((ext_vector_type(16))) __bf16 v16bf;
typedef __attribute__((ext_vector_type(8)))  float  v8f;

#define B_   8
#define S_   1024
#define D_   1024
#define H_   8
#define DH_  128
#define NEG_FILL_F (-4294967295.0f)
#define SCALE_W    0.70710678118654752440f
#define EPS_       1e-5f

union FragAB { v16bf v; unsigned int u[8]; };
union FragC  { v8f   v; float        f[8]; };

// fp32 -> bf16 round-to-nearest-even
__device__ __forceinline__ unsigned short f2bf(float x) {
    unsigned int u = __float_as_uint(x);
    unsigned int r = u + 0x7FFFu + ((u >> 16) & 1u);
    return (unsigned short)(r >> 16);
}

// A-matrix (16x32 bf16) K index for dword slot i, lane-half 'half'
// ISA 7.12.2: lanes 0-15: K={0..7,16..23}; lanes 16-31: K={8..15,24..31}
__device__ __forceinline__ int a_k(int i, int half) {
    return (i < 4) ? (8 * half + 2 * i) : (16 + 8 * half + 2 * (i - 4));
}

// LDS byte offset of a __shared__ object (flat LDS aperture keeps the
// in-group offset in addr[31:0])
__device__ __forceinline__ unsigned lds_off(const void* p) {
    return (unsigned)(size_t)p;
}

// async global->LDS 16-byte transfer (GVS mode: saddr base + 32-bit voffset)
#define ASYNC_B128(lds, voff, base)                                        \
    asm volatile("global_load_async_to_lds_b128 %0, %1, %2"                \
                 :: "v"(lds), "v"(voff), "s"(base) : "memory")
#define WAIT_ASYNC(n) asm volatile("s_wait_asynccnt " #n ::: "memory")

// ---------------------------------------------------------------------------
// K0: fp32 -> bf16 conversion (weights)
// ---------------------------------------------------------------------------
__global__ void cvt_f32_bf16(const float* __restrict__ in,
                             unsigned short* __restrict__ out, int n) {
    int i = blockIdx.x * blockDim.x + threadIdx.x;
    int stride = gridDim.x * blockDim.x;
    for (; i < n; i += stride) out[i] = f2bf(in[i]);
}

// ---------------------------------------------------------------------------
// K1: LayerNorm(src) -> src1 (f32) and src1b (bf16). One row per block.
// ---------------------------------------------------------------------------
__global__ void ln_in_kernel(const float* __restrict__ src,
                             const float* __restrict__ g1,
                             const float* __restrict__ b1,
                             float* __restrict__ src1f,
                             unsigned short* __restrict__ src1b) {
    const int row = blockIdx.x;
    const float* x = src + (size_t)row * D_;
    __shared__ float s1[256], s2[256];
    float sum = 0.f, sq = 0.f;
    for (int d = threadIdx.x; d < D_; d += 256) { float v = x[d]; sum += v; sq += v * v; }
    s1[threadIdx.x] = sum; s2[threadIdx.x] = sq;
    __syncthreads();
    for (int off = 128; off > 0; off >>= 1) {
        if (threadIdx.x < off) {
            s1[threadIdx.x] += s1[threadIdx.x + off];
            s2[threadIdx.x] += s2[threadIdx.x + off];
        }
        __syncthreads();
    }
    const float mean = s1[0] * (1.0f / D_);
    const float var  = s2[0] * (1.0f / D_) - mean * mean;
    const float rstd = rsqrtf(var + EPS_);
    for (int d = threadIdx.x; d < D_; d += 256) {
        float v = (x[d] - mean) * rstd * g1[d] + b1[d];
        src1f[(size_t)row * D_ + d] = v;
        src1b[(size_t)row * D_ + d] = f2bf(v);
    }
}

// ---------------------------------------------------------------------------
// K2: fused QKV GEMM, 64x64 block tile, 4 waves. Async double-buffered
// global->LDS staging (4 b128 transfers per thread-group per K step).
// ---------------------------------------------------------------------------
__global__ void __launch_bounds__(128)
gemm_qkv(const unsigned short* __restrict__ A,
         const unsigned short* __restrict__ Wq,
         const unsigned short* __restrict__ Wk,
         const unsigned short* __restrict__ Wv,
         unsigned short* __restrict__ Qb,
         unsigned short* __restrict__ Kb,
         unsigned short* __restrict__ Vb) {
    const unsigned short* W   = (blockIdx.z == 0) ? Wq : ((blockIdx.z == 1) ? Wk : Wv);
    unsigned short*       Out = (blockIdx.z == 0) ? Qb : ((blockIdx.z == 1) ? Kb : Vb);
    const int m0 = blockIdx.y * 64, n0 = blockIdx.x * 64;
    const int tid = threadIdx.x, wave = tid >> 5, lane = tid & 31;
    const int lm = lane & 15, half = lane >> 4;
    __shared__ __align__(16) unsigned int Asm[2][64 * 20];
    __shared__ __align__(16) unsigned int Bsm[2][64 * 20];

    // each thread owns two 16B chunks per tile: (row r0,quad qd), (row r0+32,qd)
    const int r0 = tid >> 2, qd = tid & 3;
    const int r1 = r0 + 32;
    const unsigned bufStride = 64 * 20 * 4;
    const unsigned aL0 = lds_off(&Asm[0][0]) + (r0 * 20 + qd * 4) * 4;
    const unsigned aL1 = lds_off(&Asm[0][0]) + (r1 * 20 + qd * 4) * 4;
    const unsigned bL0 = lds_off(&Bsm[0][0]) + (r0 * 20 + qd * 4) * 4;
    const unsigned bL1 = lds_off(&Bsm[0][0]) + (r1 * 20 + qd * 4) * 4;
    const unsigned aO0 = (unsigned)((m0 + r0) * D_) * 2u + qd * 16u;
    const unsigned aO1 = (unsigned)((m0 + r1) * D_) * 2u + qd * 16u;
    const unsigned bO0 = (unsigned)((n0 + r0) * D_) * 2u + qd * 16u;
    const unsigned bO1 = (unsigned)((n0 + r1) * D_) * 2u + qd * 16u;

    FragC acc[4];
#pragma unroll
    for (int s = 0; s < 4; s++)
#pragma unroll
        for (int r = 0; r < 8; r++) acc[s].f[r] = 0.f;

    // preload K-step 0 into buffer 0
    ASYNC_B128(aL0, aO0, A);
    ASYNC_B128(aL1, aO1, A);
    ASYNC_B128(bL0, bO0, W);
    ASYNC_B128(bL1, bO1, W);

    for (int kt = 0; kt < 32; ++kt) {
        const int buf = kt & 1;
        if (kt + 1 < 32) {
            const unsigned ko = (unsigned)(kt + 1) * 64u;   // 32 bf16 = 64 bytes
            const unsigned l  = (unsigned)(buf ^ 1) * bufStride;
            ASYNC_B128(aL0 + l, aO0 + ko, A);
            ASYNC_B128(aL1 + l, aO1 + ko, A);
            ASYNC_B128(bL0 + l, bO0 + ko, W);
            ASYNC_B128(bL1 + l, bO1 + ko, W);
            WAIT_ASYNC(0x4);   // newest 4 may remain; current tile resident
        } else {
            WAIT_ASYNC(0x0);
        }
        __syncthreads();

        const unsigned int* As = &Asm[buf][0];
        const unsigned int* Bs = &Bsm[buf][0];
        FragAB a;
#pragma unroll
        for (int i = 0; i < 8; i++)
            a.u[i] = As[(wave * 16 + lm) * 20 + half * 8 + i];
#pragma unroll
        for (int s = 0; s < 4; s++) {
            FragAB b;
#pragma unroll
            for (int i = 0; i < 8; i++)
                b.u[i] = Bs[(s * 16 + lm) * 20 + half * 8 + i];
            acc[s].v = __builtin_amdgcn_wmma_f32_16x16x32_bf16(
                false, a.v, false, b.v, (short)0, acc[s].v, false, false);
        }
        __syncthreads();   // all waves done reading before buffer reuse
    }
#pragma unroll
    for (int s = 0; s < 4; s++)
#pragma unroll
        for (int r = 0; r < 8; r++) {
            int i = m0 + wave * 16 + r + 8 * half;
            int j = n0 + s * 16 + lm;
            Out[(size_t)i * D_ + j] = f2bf(acc[s].f[r]);
        }
}

// ---------------------------------------------------------------------------
// K3: attention. One block = one (b, h, 16-query-row tile). 4 waves.
// Scores via WMMA (K-frags as coalesced b128), softmax in LDS, attn out,
// then attn @ V with cooperative transposed V staging in LDS.
// ---------------------------------------------------------------------------
__global__ void __launch_bounds__(128)
attn_kernel(const unsigned short* __restrict__ Qb,
            const unsigned short* __restrict__ Kb,
            const unsigned short* __restrict__ Vb,
            const unsigned char* __restrict__ mask,
            const float* __restrict__ qmask,
            const float* __restrict__ gfp,
            float* __restrict__ attn_out,
            unsigned short* __restrict__ resb) {
    extern __shared__ __align__(16) char smem[];
    float*        scores = (float*)smem;                                  // 16x1024 f32
    unsigned int* qtile  = (unsigned int*)(smem + 65536);                 // 16x64 dwords
    float*        red    = (float*)(smem + 69632);                        // 256 f32
    unsigned int* vt     = (unsigned int*)(smem + 70656);                 // 128x17 dwords

    const int bid = blockIdx.x;
    const int b   = bid / (H_ * 64);
    const int rem = bid % (H_ * 64);
    const int h   = rem / 64;
    const int q0  = (rem % 64) * 16;
    const int tid = threadIdx.x, wave = tid >> 5, lane = tid & 31;
    const int lm = lane & 15, half = lane >> 4;
    const float invgf  = 1.0f / gfp[0];
    const float rscale = 0.08838834764831845f; // 1/sqrt(128)

    // stage Q tile (16 rows x 128 d bf16) via coalesced b128 loads
#pragma unroll
    for (int it = 0; it < 2; ++it) {
        int c = tid + it * 128;              // 256 quad-chunks
        int row = c >> 4, quad = c & 15;
        const unsigned short* qp =
            Qb + (size_t)(b * S_ + q0 + row) * D_ + h * DH_ + quad * 8;
        *(uint4*)&qtile[row * 64 + quad * 4] = *(const uint4*)qp;
    }
    __syncthreads();

    // ---- scores: Q (16x128) . K^T per 16-key tile ----
    for (int j = wave; j < 64; j += 4) {
        FragC acc;
#pragma unroll
        for (int r = 0; r < 8; r++) acc.f[r] = 0.f;
#pragma unroll
        for (int c = 0; c < 4; c++) {
            FragAB a, bf;
#pragma unroll
            for (int i = 0; i < 8; i++)
                a.u[i] = qtile[lm * 64 + c * 16 + half * 8 + i];
            const int key = j * 16 + lm;
            const unsigned short* kp =
                Kb + (size_t)(b * S_ + key) * D_ + h * DH_ + c * 32 + half * 16;
            uint4 k0 = *(const uint4*)kp;
            uint4 k1 = *(const uint4*)(kp + 8);
            bf.u[0] = k0.x; bf.u[1] = k0.y; bf.u[2] = k0.z; bf.u[3] = k0.w;
            bf.u[4] = k1.x; bf.u[5] = k1.y; bf.u[6] = k1.z; bf.u[7] = k1.w;
            acc.v = __builtin_amdgcn_wmma_f32_16x16x32_bf16(
                false, a.v, false, bf.v, (short)0, acc.v, false, false);
        }
#pragma unroll
        for (int r = 0; r < 8; r++) {
            const int m   = r + 8 * half;
            const int q   = q0 + m;
            const int key = j * 16 + lm;
            const float diff = (float)(q - key);
            float val = acc.f[r] * rscale - diff * diff * invgf;
            if (mask[((size_t)(b * S_) + q) * S_ + key]) val = NEG_FILL_F;
            scores[m * S_ + key] = val;
        }
    }
    __syncthreads();

    // ---- softmax over 16 rows (8 threads/row) ----
    {
        const int row = tid >> 3, seg = tid & 7;
        float* sr = scores + row * S_;
        float mx = -3.4e38f;
        for (int c = seg * 128; c < seg * 128 + 128; c++) mx = fmaxf(mx, sr[c]);
        red[row * 8 + seg] = mx;
        __syncthreads();
        mx = red[row * 8];
        for (int t = 1; t < 8; t++) mx = fmaxf(mx, red[row * 8 + t]);
        float sum = 0.f;
        for (int c = seg * 128; c < seg * 128 + 128; c++) {
            float e = __expf(sr[c] - mx);
            sr[c] = e;
            sum += e;
        }
        red[128 + row * 8 + seg] = sum;
        __syncthreads();
        sum = 0.f;
        for (int t = 0; t < 8; t++) sum += red[128 + row * 8 + t];
        const float qm    = qmask[b * S_ + q0 + row];
        const float scale = qm / sum;
        float* arow = attn_out + (size_t)((h * B_ + b) * S_ + q0 + row) * S_;
        for (int c = seg * 128; c < seg * 128 + 128; c++) {
            float p = sr[c] * scale;
            sr[c] = p;
            arow[c] = p;
        }
    }
    __syncthreads();

    // ---- attn (16x1024) @ V (1024x128): V staged transposed per 32-key chunk
    const int s0 = wave * 2;
    const int pr   = tid >> 3;           // key-pair 0..15
    const int dseg = (tid & 7) * 16;     // d segment base
    const unsigned short* vbase = Vb + (size_t)(b * S_) * D_ + h * DH_;
    FragC acc0, acc1;
#pragma unroll
    for (int r = 0; r < 8; r++) { acc0.f[r] = 0.f; acc1.f[r] = 0.f; }

    for (int kc = 0; kc < 32; ++kc) {
        // stage Vt[d][key-pair] for keys kc*32..kc*32+31 (coalesced reads)
        const unsigned short* rp = vbase + (size_t)(kc * 32 + 2 * pr) * D_ + dseg;
        unsigned u0[8], u1[8];
        *(uint4*)&u0[0] = *(const uint4*)rp;
        *(uint4*)&u0[4] = *(const uint4*)(rp + 8);
        *(uint4*)&u1[0] = *(const uint4*)(rp + D_);
        *(uint4*)&u1[4] = *(const uint4*)(rp + D_ + 8);
#pragma unroll
        for (int j = 0; j < 16; ++j) {
            unsigned lo = (u0[j >> 1] >> (16 * (j & 1))) & 0xffffu;
            unsigned hi = (u1[j >> 1] >> (16 * (j & 1))) & 0xffffu;
            vt[(dseg + j) * 17 + pr] = lo | (hi << 16);
        }
        __syncthreads();

        FragAB a;
#pragma unroll
        for (int i = 0; i < 8; i++) {
            const int k = kc * 32 + 16 * half + 2 * i;
            const unsigned int lo = f2bf(scores[lm * S_ + k]);
            const unsigned int hi = f2bf(scores[lm * S_ + k + 1]);
            a.u[i] = lo | (hi << 16);
        }
#pragma unroll
        for (int ss = 0; ss < 2; ++ss) {
            const int dcol = (s0 + ss) * 16 + lm;
            FragAB bfv;
#pragma unroll
            for (int i = 0; i < 8; i++)
                bfv.u[i] = vt[dcol * 17 + half * 8 + i];
            if (ss == 0)
                acc0.v = __builtin_amdgcn_wmma_f32_16x16x32_bf16(
                    false, a.v, false, bfv.v, (short)0, acc0.v, false, false);
            else
                acc1.v = __builtin_amdgcn_wmma_f32_16x16x32_bf16(
                    false, a.v, false, bfv.v, (short)0, acc1.v, false, false);
        }
        __syncthreads();
    }
#pragma unroll
    for (int r = 0; r < 8; r++) {
        const int q = q0 + r + 8 * half;
        resb[(size_t)(b * S_ + q) * D_ + h * DH_ + s0 * 16 + lm]       = f2bf(acc0.f[r]);
        resb[(size_t)(b * S_ + q) * D_ + h * DH_ + (s0 + 1) * 16 + lm] = f2bf(acc1.f[r]);
    }
}

// ---------------------------------------------------------------------------
// K4a: output projection. proj = concat(src1, res) @ Wo.T + bo. K = 2048,
// async double-buffered staging; A source switches halfway through K.
// ---------------------------------------------------------------------------
__global__ void __launch_bounds__(128)
gemm_proj(const unsigned short* __restrict__ A1,
          const unsigned short* __restrict__ A2,
          const unsigned short* __restrict__ Wob,
          const float* __restrict__ bo,
          float* __restrict__ proj) {
    const int m0 = blockIdx.y * 64, n0 = blockIdx.x * 64;
    const int tid = threadIdx.x, wave = tid >> 5, lane = tid & 31;
    const int lm = lane & 15, half = lane >> 4;
    __shared__ __align__(16) unsigned int Asm[2][64 * 20];
    __shared__ __align__(16) unsigned int Bsm[2][64 * 20];

    const int r0 = tid >> 2, qd = tid & 3;
    const int r1 = r0 + 32;
    const unsigned bufStride = 64 * 20 * 4;
    const unsigned aL0 = lds_off(&Asm[0][0]) + (r0 * 20 + qd * 4) * 4;
    const unsigned aL1 = lds_off(&Asm[0][0]) + (r1 * 20 + qd * 4) * 4;
    const unsigned bL0 = lds_off(&Bsm[0][0]) + (r0 * 20 + qd * 4) * 4;
    const unsigned bL1 = lds_off(&Bsm[0][0]) + (r1 * 20 + qd * 4) * 4;
    const unsigned aO0 = (unsigned)((m0 + r0) * D_) * 2u + qd * 16u;
    const unsigned aO1 = (unsigned)((m0 + r1) * D_) * 2u + qd * 16u;
    const unsigned bO0 = (unsigned)((n0 + r0) * (2 * D_)) * 2u + qd * 16u;
    const unsigned bO1 = (unsigned)((n0 + r1) * (2 * D_)) * 2u + qd * 16u;

    auto issue = [&](int kt, unsigned l) {
        const unsigned short* Ab = (kt < 32) ? A1 : A2;
        const unsigned ka = (unsigned)((kt < 32) ? kt : kt - 32) * 64u;
        const unsigned kb = (unsigned)kt * 64u;
        ASYNC_B128(aL0 + l, aO0 + ka, Ab);
        ASYNC_B128(aL1 + l, aO1 + ka, Ab);
        ASYNC_B128(bL0 + l, bO0 + kb, Wob);
        ASYNC_B128(bL1 + l, bO1 + kb, Wob);
    };

    FragC acc[4];
#pragma unroll
    for (int s = 0; s < 4; s++)
#pragma unroll
        for (int r = 0; r < 8; r++) acc[s].f[r] = 0.f;

    issue(0, 0);
    for (int kt = 0; kt < 64; ++kt) {
        const int buf = kt & 1;
        if (kt + 1 < 64) {
            issue(kt + 1, (unsigned)(buf ^ 1) * bufStride);
            WAIT_ASYNC(0x4);
        } else {
            WAIT_ASYNC(0x0);
        }
        __syncthreads();

        const unsigned int* As = &Asm[buf][0];
        const unsigned int* Bs = &Bsm[buf][0];
        FragAB a;
#pragma unroll
        for (int i = 0; i < 8; i++)
            a.u[i] = As[(wave * 16 + lm) * 20 + half * 8 + i];
#pragma unroll
        for (int s = 0; s < 4; s++) {
            FragAB b;
#pragma unroll
            for (int i = 0; i < 8; i++)
                b.u[i] = Bs[(s * 16 + lm) * 20 + half * 8 + i];
            acc[s].v = __builtin_amdgcn_wmma_f32_16x16x32_bf16(
                false, a.v, false, b.v, (short)0, acc[s].v, false, false);
        }
        __syncthreads();
    }
#pragma unroll
    for (int s = 0; s < 4; s++)
#pragma unroll
        for (int r = 0; r < 8; r++) {
            int i = m0 + wave * 16 + r + 8 * half;
            int j = n0 + s * 16 + lm;
            proj[(size_t)i * D_ + j] = acc[s].f[r] + bo[j];
        }
}

// ---------------------------------------------------------------------------
// K4b: out = (src + LayerNorm(proj + src1; g2,b2)) * sqrt(1/2)
// ---------------------------------------------------------------------------
__global__ void ln_out_kernel(const float* __restrict__ proj,
                              const float* __restrict__ src1f,
                              const float* __restrict__ src,
                              const float* __restrict__ g2,
                              const float* __restrict__ b2,
                              float* __restrict__ out) {
    const int row = blockIdx.x;
    const float* p  = proj  + (size_t)row * D_;
    const float* s1 = src1f + (size_t)row * D_;
    __shared__ float r1[256], r2[256];
    float sum = 0.f, sq = 0.f;
    for (int d = threadIdx.x; d < D_; d += 256) {
        float v = p[d] + s1[d];
        sum += v; sq += v * v;
    }
    r1[threadIdx.x] = sum; r2[threadIdx.x] = sq;
    __syncthreads();
    for (int off = 128; off > 0; off >>= 1) {
        if (threadIdx.x < off) {
            r1[threadIdx.x] += r1[threadIdx.x + off];
            r2[threadIdx.x] += r2[threadIdx.x + off];
        }
        __syncthreads();
    }
    const float mean = r1[0] * (1.0f / D_);
    const float var  = r2[0] * (1.0f / D_) - mean * mean;
    const float rstd = rsqrtf(var + EPS_);
    for (int d = threadIdx.x; d < D_; d += 256) {
        float y  = p[d] + s1[d];
        float ln = (y - mean) * rstd * g2[d] + b2[d];
        out[(size_t)row * D_ + d] = (src[(size_t)row * D_ + d] + ln) * SCALE_W;
    }
}

// ---------------------------------------------------------------------------
// kernel_launch
// ---------------------------------------------------------------------------
extern "C" void kernel_launch(void* const* d_in, const int* in_sizes, int n_in,
                              void* d_out, int out_size, void* d_ws, size_t ws_size,
                              hipStream_t stream) {
    const float*         src   = (const float*)d_in[0];
    const unsigned char* mask  = (const unsigned char*)d_in[1];
    const float*         qmask = (const float*)d_in[2];
    const float*         g1    = (const float*)d_in[3];
    const float*         b1    = (const float*)d_in[4];
    const float*         Wq    = (const float*)d_in[5];
    const float*         Wk    = (const float*)d_in[6];
    const float*         Wv    = (const float*)d_in[7];
    const float*         Wo    = (const float*)d_in[8];
    const float*         bo    = (const float*)d_in[9];
    const float*         g2    = (const float*)d_in[10];
    const float*         b2    = (const float*)d_in[11];
    const float*         gf    = (const float*)d_in[12];

    char* ws = (char*)d_ws;
    size_t off = 0;
    auto alloc = [&](size_t bytes) -> char* {
        char* p = ws + off;
        off += (bytes + 255) & ~(size_t)255;
        return p;
    };
    const size_t BSD = (size_t)B_ * S_ * D_;
    float*          src1f = (float*)alloc(BSD * 4);
    unsigned short* src1b = (unsigned short*)alloc(BSD * 2);
    unsigned short* Wqb   = (unsigned short*)alloc((size_t)D_ * D_ * 2);
    unsigned short* Wkb   = (unsigned short*)alloc((size_t)D_ * D_ * 2);
    unsigned short* Wvb   = (unsigned short*)alloc((size_t)D_ * D_ * 2);
    unsigned short* Wob   = (unsigned short*)alloc((size_t)D_ * 2 * D_ * 2);
    unsigned short* Qb    = (unsigned short*)alloc(BSD * 2);
    unsigned short* Kb    = (unsigned short*)alloc(BSD * 2);
    unsigned short* Vb    = (unsigned short*)alloc(BSD * 2);
    unsigned short* resb  = (unsigned short*)alloc(BSD * 2);
    float*          proj  = (float*)alloc(BSD * 4);

    float* out      = (float*)d_out;
    float* attn_out = out + BSD;   // [H*B, S, S] follows [B, S, D]

    cvt_f32_bf16<<<2048, 256, 0, stream>>>(Wq, Wqb, D_ * D_);
    cvt_f32_bf16<<<2048, 256, 0, stream>>>(Wk, Wkb, D_ * D_);
    cvt_f32_bf16<<<2048, 256, 0, stream>>>(Wv, Wvb, D_ * D_);
    cvt_f32_bf16<<<2048, 256, 0, stream>>>(Wo, Wob, D_ * 2 * D_);

    ln_in_kernel<<<B_ * S_, 256, 0, stream>>>(src, g1, b1, src1f, src1b);

    gemm_qkv<<<dim3(D_ / 64, (B_ * S_) / 64, 3), 128, 0, stream>>>(
        src1b, Wqb, Wkb, Wvb, Qb, Kb, Vb);

    const size_t attn_lds = 65536 + 4096 + 1024 + 128 * 17 * 4;   // 79360
    attn_kernel<<<B_ * H_ * (S_ / 16), 128, attn_lds, stream>>>(
        Qb, Kb, Vb, mask, qmask, gf, attn_out, resb);

    gemm_proj<<<dim3(D_ / 64, (B_ * S_) / 64), 128, 0, stream>>>(
        src1b, resb, Wob, bo, proj);

    ln_out_kernel<<<B_ * S_, 256, 0, stream>>>(proj, src1f, src, g2, b2, out);
}
